// Encoder_dpm_41283225649648
// MI455X (gfx1250) — matrix operand order repaired and checked by reading, not compiled, not run on hardware
//
#include <hip/hip_runtime.h>
#include <hip/hip_bf16.h>

#define NN 25000
#define NE 400000
#define NPAD 25008      // 1563 * 16
#define NTILES 1563
#define TILE_STRIDE 136 // 128 + 8 halves pad -> conflict-free LDS b128 reads

typedef __attribute__((ext_vector_type(16))) _Float16 v16h;
typedef __attribute__((ext_vector_type(8)))  _Float16 v8h;
typedef __attribute__((ext_vector_type(8)))  float    v8f;
typedef __attribute__((ext_vector_type(4)))  float    v4f;

// ---------------- device helpers ----------------

__device__ __forceinline__ v16h combine8(v8h lo, v8h hi) {
  return __builtin_shufflevector(lo, hi, 0,1,2,3,4,5,6,7,8,9,10,11,12,13,14,15);
}

__device__ __forceinline__ v8f wmma(v16h a, v16h b, v8f c) {
  return __builtin_amdgcn_wmma_f32_16x16x32_f16(false, a, false, b, (short)0, c, false, false);
}

__device__ __forceinline__ v8f bsplat(float b) {
  v8f v;
#pragma unroll
  for (int q = 0; q < 8; ++q) v[q] = b;
  return v;
}

// A chunk from f16 row; row_h = row_ptr + hlf*8 precomputed. kb is a constant.
__device__ __forceinline__ v16h load_a_f16(const _Float16* row_h, int kb) {
  v8h lo = *(const v8h*)(row_h + kb*32);
  v8h hi = *(const v8h*)(row_h + kb*32 + 16);
  return combine8(lo, hi);
}

// A chunk from f32 row (convert to f16); hi 8-run optionally zero.
__device__ __forceinline__ v16h load_a_f32c(const float* row_h, int kb, bool hi_valid) {
  v4f x0 = *(const v4f*)(row_h + kb*32);
  v4f x1 = *(const v4f*)(row_h + kb*32 + 4);
  v4f x2 = {}, x3 = {};
  if (hi_valid) {
    x2 = *(const v4f*)(row_h + kb*32 + 16);
    x3 = *(const v4f*)(row_h + kb*32 + 20);
  }
  v16h a;
#pragma unroll
  for (int q = 0; q < 4; ++q) {
    a[q]    = (_Float16)x0[q];
    a[4+q]  = (_Float16)x1[q];
    a[8+q]  = (_Float16)x2[q];
    a[12+q] = (_Float16)x3[q];
  }
  return a;
}

// B operand: wp = weight_base + lane*16 precomputed; (kb,nb) constant -> offset.
// Works for both global (swizzled weights) and LDS-staged copies.
__device__ __forceinline__ v16h load_bp(const _Float16* wp, int kb, int nb) {
  return *(const v16h*)(wp + (size_t)(kb*8 + nb)*512);
}

__device__ __forceinline__ float silu(float x) {
  return __fdividef(x, 1.f + __expf(-x));
}

// SiLU(acc) -> f16 into wave-private LDS tile (bias already folded into acc).
// tw = tile + hlf*8*TILE_STRIDE + lcol  (precomputed base)
__device__ __forceinline__ void d_to_tile_silu(_Float16* tw, const v8f* acc) {
#pragma unroll
  for (int r = 0; r < 8; ++r) {
#pragma unroll
    for (int nb = 0; nb < 8; ++nb) {
      tw[r*TILE_STRIDE + nb*16] = (_Float16)silu(acc[nb][r]);
    }
  }
}

// Read A chunk from tile; ta = tile + lcol*TILE_STRIDE + hlf*8 (precomputed base)
__device__ __forceinline__ v16h tile_a(const _Float16* ta, int kb) {
  v8h lo = *(const v8h*)(ta + kb*32);
  v8h hi = *(const v8h*)(ta + kb*32 + 16);
  return combine8(lo, hi);
}

// ---------------- prep kernels ----------------

// Swizzle fp32 weight [K x 128] into WMMA-B layout (dense B: k = kb*32 + (idx<8 ? (lane/16)*8 + idx : 16 + (lane/16)*8 + idx-8))
__global__ void swizzle_weight(const float* __restrict__ W, _Float16* __restrict__ dst,
                               int K, int total) {
  int tid = blockIdx.x * blockDim.x + threadIdx.x;
  if (tid >= total) return;
  int idx  = tid & 15;
  int lane = (tid >> 4) & 31;
  int nb   = (tid >> 9) & 7;
  int kb   = tid >> 12;
  int n = nb*16 + (lane & 15);
  int k = kb*32 + (idx < 8 ? (lane >> 4)*8 + idx : 16 + (lane >> 4)*8 + (idx - 8));
  float v = (k < K) ? W[(size_t)k*128 + n] : 0.f;
  dst[tid] = (_Float16)v;
}

__global__ void zero_kernel(float* __restrict__ agg, float* __restrict__ v0,
                            long long na, long long nv) {
  long long i = (long long)blockIdx.x * blockDim.x + threadIdx.x;
  long long stride = (long long)gridDim.x * blockDim.x;
  for (long long k = i; k < na + nv; k += stride) {
    if (k < na) agg[k] = 0.f; else v0[k - na] = 0.f;
  }
}

// ---------------- node kernel 1: f = species @ Wa + ba ----------------

__global__ __launch_bounds__(256) void node_f_kernel(
    const float* __restrict__ species, const _Float16* __restrict__ was,
    const float* __restrict__ ba, float* __restrict__ f32f, _Float16* __restrict__ f16f)
{
  const int lane = threadIdx.x & 31, wave = threadIdx.x >> 5;
  const int hlf = lane >> 4, lcol = lane & 15;
  const _Float16* wap = was + lane*16;
  float bal[8];
#pragma unroll
  for (int nb = 0; nb < 8; ++nb) bal[nb] = ba[nb*16 + lcol];
  const int gw = blockIdx.x*8 + wave, nw = gridDim.x*8;
  for (int tl = gw; tl < NTILES; tl += nw) {
    int rbase = tl*16;
    int arow = rbase + lcol;
    bool av = arow < NN;
    const float* row = species + (unsigned)arow*100u;
    const float* row_h = row + hlf*8;
    v8f acc[8];
#pragma unroll
    for (int nb = 0; nb < 8; ++nb) acc[nb] = bsplat(bal[nb]);   // bias via C operand
#pragma unroll
    for (int kb = 0; kb < 4; ++kb) {
      v16h a = {};
      if (kb < 3) {
        if (av) a = load_a_f32c(row_h, kb, true);
      } else {
        // k = 96 + hlf*8 + q : only hlf==0, q<4 valid (k=96..99)
        if (av && hlf == 0) {
          v4f x = *(const v4f*)(row + 96);
#pragma unroll
          for (int q = 0; q < 4; ++q) a[q] = (_Float16)x[q];
        }
      }
#pragma unroll
      for (int nb = 0; nb < 8; ++nb) acc[nb] = wmma(a, load_bp(wap, kb, nb), acc[nb]);
    }
#pragma unroll
    for (int r = 0; r < 8; ++r) {
      unsigned off = (unsigned)(rbase + hlf*8 + r)*128u + lcol;
      float* p32 = f32f + off;
      _Float16* p16 = f16f + off;
#pragma unroll
      for (int nb = 0; nb < 8; ++nb) {
        float x = acc[nb][r];
        p32[nb*16] = x;
        p16[nb*16] = (_Float16)x;
      }
    }
  }
}

// ---------------- fused edge kernel ----------------
// LDS: 256 KB of staged weights (Ws1|Wv1|Ws2|Wv2) + 34.8 KB relayout tiles.
// CDNA5 WGP has 320 KB LDS; ~400-VGPR waves mean ~1 workgroup/WGP anyway.

#define SW_WS1 0
#define SW_WV1 49152
#define SW_WS2 98304
#define SW_WV2 114688
#define SW_TOTAL 131072   // halves

__global__ __launch_bounds__(256) void edge_kernel(
    const int* __restrict__ eidx32,      // low dwords of int64 edge_index
    const float* __restrict__ edge_attr, const float* __restrict__ edge_vec,
    const float* __restrict__ f32f, const _Float16* __restrict__ f16f,
    const _Float16* __restrict__ wb1s, const float* __restrict__ bb1,
    const _Float16* __restrict__ wb2s, const float* __restrict__ bb2,
    const _Float16* __restrict__ ws1s, const float* __restrict__ bs1,
    const _Float16* __restrict__ ws2s, const float* __restrict__ bs2,
    const _Float16* __restrict__ wv1s, const float* __restrict__ bv1,
    const _Float16* __restrict__ wv2s, const float* __restrict__ bv2,
    float* __restrict__ ea_out, float* __restrict__ agg, float* __restrict__ v0)
{
  __shared__ __align__(32) _Float16 sW[SW_TOTAL];
  __shared__ __align__(32) _Float16 ldsT[8 * 16 * TILE_STRIDE];

  // ---- stage hot weights into LDS (once per workgroup) ----
  {
    const int tid = threadIdx.x;
    v8h* d = (v8h*)sW;
    const v8h* s1 = (const v8h*)ws1s;
    const v8h* v1 = (const v8h*)wv1s;
    const v8h* s2 = (const v8h*)ws2s;
    const v8h* v2 = (const v8h*)wv2s;
    for (int o = tid; o < 49152/8; o += 256) d[SW_WS1/8 + o] = s1[o];
    for (int o = tid; o < 49152/8; o += 256) d[SW_WV1/8 + o] = v1[o];
    for (int o = tid; o < 16384/8; o += 256) d[SW_WS2/8 + o] = s2[o];
    for (int o = tid; o < 16384/8; o += 256) d[SW_WV2/8 + o] = v2[o];
  }
  __syncthreads();

  const int lane = threadIdx.x & 31, wave = threadIdx.x >> 5;
  _Float16* tile = &ldsT[wave * 16 * TILE_STRIDE];
  const int hlf = lane >> 4, lcol = lane & 15;
  _Float16* tw = tile + hlf*8*TILE_STRIDE + lcol;       // D->tile write base
  const _Float16* ta = tile + lcol*TILE_STRIDE + hlf*8; // tile->A read base

  const _Float16* wb1p = wb1s + lane*16;                // from L2
  const _Float16* wb2p = wb2s + lane*16;                // from L2
  const _Float16* ws1p = sW + SW_WS1 + lane*16;         // from LDS
  const _Float16* ws2p = sW + SW_WS2 + lane*16;
  const _Float16* wv1p = sW + SW_WV1 + lane*16;
  const _Float16* wv2p = sW + SW_WV2 + lane*16;

  float bb1l[8], bb2l[8], bs1l[8], bs2l[8], bv1l[8], bv2l[8];
#pragma unroll
  for (int nb = 0; nb < 8; ++nb) {
    int c = nb*16 + lcol;
    bb1l[nb]=bb1[c]; bb2l[nb]=bb2[c]; bs1l[nb]=bs1[c];
    bs2l[nb]=bs2[c]; bv1l[nb]=bv1[c]; bv2l[nb]=bv2[c];
  }

  const int gw = blockIdx.x*8 + wave, nw = gridDim.x*8;
  for (int wt = gw; wt < NE/32; wt += nw) {
    const int ebase = wt*32;
    // A-gather row pointers (row = lane%16), shared by msg & vec MLPs
    const _Float16 *pAi[2], *pAj[2];
#pragma unroll
    for (int t = 0; t < 2; ++t) {
      int e = ebase + t*16 + lcol;
      unsigned i = (unsigned)eidx32[2*e];
      unsigned j = (unsigned)eidx32[2*(NE + e)];
      pAi[t] = f16f + i*128u + hlf*8;
      pAj[t] = f16f + j*128u + hlf*8;
    }
    // D-row indices (row = r + hlf*8)
    int dI[2][8], dJ[2][8];
#pragma unroll
    for (int t = 0; t < 2; ++t)
#pragma unroll
      for (int r = 0; r < 8; ++r) {
        int e = ebase + t*16 + hlf*8 + r;
        dI[t][r] = eidx32[2*e];
        dJ[t][r] = eidx32[2*(NE + e)];
      }

    v8f acc[2][8];
    // ===== ea layer 1: edge_attr(120) @ Wb1 + bb1 =====
#pragma unroll
    for (int t = 0; t < 2; ++t)
#pragma unroll
      for (int nb = 0; nb < 8; ++nb) acc[t][nb] = bsplat(bb1l[nb]);
    {
      const float* rA[2];
#pragma unroll
      for (int t = 0; t < 2; ++t)
        rA[t] = edge_attr + (unsigned)(ebase + t*16 + lcol)*120u + hlf*8;
#pragma unroll
      for (int kb = 0; kb < 4; ++kb) {
        bool hiv = (kb < 3) || (hlf == 0);  // 120 cols: hi 8-run of kb=3,hlf=1 is pad
        v16h a[2];
#pragma unroll
        for (int t = 0; t < 2; ++t) a[t] = load_a_f32c(rA[t], kb, hiv);
#pragma unroll
        for (int nb = 0; nb < 8; ++nb) {
          v16h b = load_bp(wb1p, kb, nb);
#pragma unroll
          for (int t = 0; t < 2; ++t) acc[t][nb] = wmma(a[t], b, acc[t][nb]);
        }
      }
    }
    v16h hA[2][4];
#pragma unroll
    for (int t = 0; t < 2; ++t) {
      d_to_tile_silu(tw, acc[t]);
#pragma unroll
      for (int kb = 0; kb < 4; ++kb) hA[t][kb] = tile_a(ta, kb);
    }
    // ===== ea layer 2 (+bb2 via C) =====
#pragma unroll
    for (int t = 0; t < 2; ++t)
#pragma unroll
      for (int nb = 0; nb < 8; ++nb) acc[t][nb] = bsplat(bb2l[nb]);
#pragma unroll
    for (int kb = 0; kb < 4; ++kb)
#pragma unroll
      for (int nb = 0; nb < 8; ++nb) {
        v16h b = load_bp(wb2p, kb, nb);
#pragma unroll
        for (int t = 0; t < 2; ++t) acc[t][nb] = wmma(hA[t][kb], b, acc[t][nb]);
      }
    v16h eA[2][4];
#pragma unroll
    for (int t = 0; t < 2; ++t) {
#pragma unroll
      for (int r = 0; r < 8; ++r) {
        unsigned e = (unsigned)(ebase + t*16 + hlf*8 + r);
        float* pea = ea_out + e*128u + lcol;
        _Float16* pt = tw + r*TILE_STRIDE;
#pragma unroll
        for (int nb = 0; nb < 8; ++nb) {
          float x = acc[t][nb][r];
          pea[nb*16] = x;               // fp32 output
          pt[nb*16] = (_Float16)x;      // relayout via LDS
        }
      }
#pragma unroll
      for (int kb = 0; kb < 4; ++kb) eA[t][kb] = tile_a(ta, kb);
    }

    // ===== fused layer-1 for both MLPs: [fi|fj|ea] @ {Ws1, Wv1} (B from LDS) =====
    v8f accV[2][8];
#pragma unroll
    for (int t = 0; t < 2; ++t)
#pragma unroll
      for (int nb = 0; nb < 8; ++nb) {
        acc[t][nb]  = bsplat(bs1l[nb]);
        accV[t][nb] = bsplat(bv1l[nb]);
      }
#pragma unroll
    for (int kb = 0; kb < 12; ++kb) {
      v16h a[2];
#pragma unroll
      for (int t = 0; t < 2; ++t) {
        if (kb < 4)      a[t] = load_a_f16(pAi[t], kb);
        else if (kb < 8) a[t] = load_a_f16(pAj[t], kb-4);
        else             a[t] = eA[t][kb-8];
      }
#pragma unroll
      for (int nb = 0; nb < 8; ++nb) {
        v16h bS = load_bp(ws1p, kb, nb);
#pragma unroll
        for (int t = 0; t < 2; ++t) acc[t][nb] = wmma(a[t], bS, acc[t][nb]);
        v16h bV = load_bp(wv1p, kb, nb);
#pragma unroll
        for (int t = 0; t < 2; ++t) accV[t][nb] = wmma(a[t], bV, accV[t][nb]);
      }
    }

    // ===== message layer 2 (+bs2), scale by f[i], scatter into agg[j] =====
#pragma unroll
    for (int t = 0; t < 2; ++t) {
      d_to_tile_silu(tw, acc[t]);
#pragma unroll
      for (int kb = 0; kb < 4; ++kb) hA[t][kb] = tile_a(ta, kb);
    }
#pragma unroll
    for (int t = 0; t < 2; ++t)
#pragma unroll
      for (int nb = 0; nb < 8; ++nb) acc[t][nb] = bsplat(bs2l[nb]);
#pragma unroll
    for (int kb = 0; kb < 4; ++kb)
#pragma unroll
      for (int nb = 0; nb < 8; ++nb) {
        v16h b = load_bp(ws2p, kb, nb);
#pragma unroll
        for (int t = 0; t < 2; ++t) acc[t][nb] = wmma(hA[t][kb], b, acc[t][nb]);
      }
#pragma unroll
    for (int t = 0; t < 2; ++t)
#pragma unroll
      for (int r = 0; r < 8; ++r) {
        const float* pfi = f32f + (unsigned)dI[t][r]*128u + lcol;
        float* pagg = agg + (unsigned)dJ[t][r]*128u + lcol;
#pragma unroll
        for (int nb = 0; nb < 8; ++nb) {
          atomicAdd(&pagg[nb*16], acc[t][nb][r] * pfi[nb*16]);
        }
      }

    // ===== vector layer 2 (+bv2), scatter into v0[j] =====
#pragma unroll
    for (int t = 0; t < 2; ++t) {
      d_to_tile_silu(tw, accV[t]);
#pragma unroll
      for (int kb = 0; kb < 4; ++kb) hA[t][kb] = tile_a(ta, kb);
    }
#pragma unroll
    for (int t = 0; t < 2; ++t)
#pragma unroll
      for (int nb = 0; nb < 8; ++nb) acc[t][nb] = bsplat(bv2l[nb]);
#pragma unroll
    for (int kb = 0; kb < 4; ++kb)
#pragma unroll
      for (int nb = 0; nb < 8; ++nb) {
        v16h b = load_bp(wv2p, kb, nb);
#pragma unroll
        for (int t = 0; t < 2; ++t) acc[t][nb] = wmma(hA[t][kb], b, acc[t][nb]);
      }
#pragma unroll
    for (int t = 0; t < 2; ++t)
#pragma unroll
      for (int r = 0; r < 8; ++r) {
        unsigned e = (unsigned)(ebase + t*16 + hlf*8 + r);
        const float* pev = edge_vec + e*3u;
        float e0 = pev[0], e1 = pev[1], e2 = pev[2];
        float* pv = v0 + (unsigned)dJ[t][r]*384u + (unsigned)lcol*3u;
#pragma unroll
        for (int nb = 0; nb < 8; ++nb) {
          float m = acc[t][nb][r];
          atomicAdd(&pv[nb*48 + 0], e0 * m);
          atomicAdd(&pv[nb*48 + 1], e1 * m);
          atomicAdd(&pv[nb*48 + 2], e2 * m);
        }
      }
  }
}

// ---------------- node kernel 2: h0 = MLP([f,agg]) + MLP(rff(t)) ----------------

__global__ __launch_bounds__(256) void node_h_kernel(
    const _Float16* __restrict__ f16f, const float* __restrict__ agg,
    const float* __restrict__ tin, const float* __restrict__ Brff,
    const _Float16* __restrict__ wh1s, const float* __restrict__ bh1,
    const _Float16* __restrict__ wh2s, const float* __restrict__ bh2,
    const _Float16* __restrict__ wt1s, const float* __restrict__ bt1,
    const _Float16* __restrict__ wt2s, const float* __restrict__ bt2,
    float* __restrict__ h0)
{
  __shared__ __align__(32) _Float16 lds[8 * 16 * TILE_STRIDE];
  const int lane = threadIdx.x & 31, wave = threadIdx.x >> 5;
  _Float16* tile = &lds[wave * 16 * TILE_STRIDE];
  const int hlf = lane >> 4, lcol = lane & 15;
  _Float16* tw = tile + hlf*8*TILE_STRIDE + lcol;
  const _Float16* ta = tile + lcol*TILE_STRIDE + hlf*8;
  const _Float16* wh1p = wh1s + lane*16;
  const _Float16* wh2p = wh2s + lane*16;
  const _Float16* wt1p = wt1s + lane*16;
  const _Float16* wt2p = wt2s + lane*16;
  float bh1l[8], bh2l[8], bt1l[8], bt2l[8];
#pragma unroll
  for (int nb = 0; nb < 8; ++nb) {
    int c = nb*16 + lcol;
    bh1l[nb]=bh1[c]; bh2l[nb]=bh2[c]; bt1l[nb]=bt1[c]; bt2l[nb]=bt2[c];
  }
  const int gw = blockIdx.x*8 + wave, nw = gridDim.x*8;
  for (int tl = gw; tl < NTILES; tl += nw) {
    int rbase = tl*16;
    int arow = rbase + lcol;            // < NPAD always
    const _Float16* pf = f16f + (unsigned)arow*128u + hlf*8;
    const float*    pg = agg  + (unsigned)arow*128u + hlf*8;
    v8f acc[8];
#pragma unroll
    for (int nb = 0; nb < 8; ++nb) acc[nb] = bsplat(bh1l[nb]);
    // [f | agg] @ Wh1 (K=256)
#pragma unroll
    for (int kb = 0; kb < 8; ++kb) {
      v16h a = (kb < 4) ? load_a_f16(pf, kb) : load_a_f32c(pg, kb-4, true);
#pragma unroll
      for (int nb = 0; nb < 8; ++nb) acc[nb] = wmma(a, load_bp(wh1p, kb, nb), acc[nb]);
    }
    d_to_tile_silu(tw, acc);
    v16h hA[4];
#pragma unroll
    for (int kb = 0; kb < 4; ++kb) hA[kb] = tile_a(ta, kb);
    v8f acch[8];
#pragma unroll
    for (int nb = 0; nb < 8; ++nb) acch[nb] = bsplat(bh2l[nb]);
#pragma unroll
    for (int kb = 0; kb < 4; ++kb)
#pragma unroll
      for (int nb = 0; nb < 8; ++nb)
        acch[nb] = wmma(hA[kb], load_bp(wh2p, kb, nb), acch[nb]);

    // time embedding: rff = [cos(2*pi*t*B), sin(2*pi*t*B)]
    float tv = (arow < NN) ? tin[arow] : 0.f;
#pragma unroll
    for (int nb = 0; nb < 8; ++nb) acc[nb] = bsplat(bt1l[nb]);
#pragma unroll
    for (int kb = 0; kb < 4; ++kb) {
      v16h a;
#pragma unroll
      for (int idx = 0; idx < 16; ++idx) {
        int k = kb*32 + (idx < 8 ? hlf*8 + idx : 16 + hlf*8 + (idx - 8));
        float p = 6.283185307179586f * tv * Brff[k & 63];
        a[idx] = (_Float16)((k < 64) ? cosf(p) : sinf(p));
      }
#pragma unroll
      for (int nb = 0; nb < 8; ++nb) acc[nb] = wmma(a, load_bp(wt1p, kb, nb), acc[nb]);
    }
    d_to_tile_silu(tw, acc);
#pragma unroll
    for (int kb = 0; kb < 4; ++kb) hA[kb] = tile_a(ta, kb);
    v8f acct[8];
#pragma unroll
    for (int nb = 0; nb < 8; ++nb) acct[nb] = bsplat(bt2l[nb]);
#pragma unroll
    for (int kb = 0; kb < 4; ++kb)
#pragma unroll
      for (int nb = 0; nb < 8; ++nb)
        acct[nb] = wmma(hA[kb], load_bp(wt2p, kb, nb), acct[nb]);

#pragma unroll
    for (int r = 0; r < 8; ++r) {
      int row = rbase + hlf*8 + r;
      if (row < NN) {
        float* ph = h0 + (unsigned)row*128u + lcol;
#pragma unroll
        for (int nb = 0; nb < 8; ++nb)
          ph[nb*16] = acch[nb][r] + acct[nb][r];
      }
    }
  }
}

// ---------------- host ----------------

extern "C" void kernel_launch(void* const* d_in, const int* in_sizes, int n_in,
                              void* d_out, int out_size, void* d_ws, size_t ws_size,
                              hipStream_t stream) {
  (void)in_sizes; (void)n_in; (void)out_size; (void)ws_size;
  const float* species   = (const float*)d_in[0];
  const int*   eidx32    = (const int*)d_in[1];   // low dwords of int64 indices
  const float* edge_attr = (const float*)d_in[2];
  const float* edge_vec  = (const float*)d_in[3];
  const float* tin       = (const float*)d_in[4];
  const float *Wa=(const float*)d_in[5],  *ba =(const float*)d_in[6];
  const float *Wb1=(const float*)d_in[7], *bb1=(const float*)d_in[8];
  const float *Wb2=(const float*)d_in[9], *bb2=(const float*)d_in[10];
  const float *Ws1=(const float*)d_in[11],*bs1=(const float*)d_in[12];
  const float *Ws2=(const float*)d_in[13],*bs2=(const float*)d_in[14];
  const float *Wh1=(const float*)d_in[15],*bh1=(const float*)d_in[16];
  const float *Wh2=(const float*)d_in[17],*bh2=(const float*)d_in[18];
  const float *Wv1=(const float*)d_in[19],*bv1=(const float*)d_in[20];
  const float *Wv2=(const float*)d_in[21],*bv2=(const float*)d_in[22];
  const float *Brff=(const float*)d_in[23];
  const float *Wt1=(const float*)d_in[24],*bt1=(const float*)d_in[25];
  const float *Wt2=(const float*)d_in[26],*bt2=(const float*)d_in[27];

  float* h0 = (float*)d_out;
  float* v0 = h0 + (size_t)NN*128;
  float* ea = v0 + (size_t)NN*384;

  char* p = (char*)d_ws;
  auto take = [&](size_t bytes) { char* r = p; p += (bytes + 255) & ~(size_t)255; return r; };
  float*     f32f = (float*)take((size_t)NPAD*128*4);
  float*     agg  = (float*)take((size_t)NPAD*128*4);
  _Float16*  f16f = (_Float16*)take((size_t)NPAD*128*2);
  auto takeW = [&](int Kpad) { return (_Float16*)take((size_t)Kpad*128*2); };
  _Float16 *was  = takeW(128);
  _Float16 *wb1s = takeW(128), *wb2s = takeW(128);
  _Float16 *ws1s = takeW(384), *ws2s = takeW(128);
  _Float16 *wh1s = takeW(256), *wh2s = takeW(128);
  _Float16 *wv1s = takeW(384), *wv2s = takeW(128);
  _Float16 *wt1s = takeW(128), *wt2s = takeW(128);

  auto sw = [&](const float* W, _Float16* dst, int K, int Kpad) {
    int total = Kpad * 128;
    swizzle_weight<<<total/256, 256, 0, stream>>>(W, dst, K, total);
  };
  sw(Wa,  was,  100, 128);
  sw(Wb1, wb1s, 120, 128); sw(Wb2, wb2s, 128, 128);
  sw(Ws1, ws1s, 384, 384); sw(Ws2, ws2s, 128, 128);
  sw(Wh1, wh1s, 256, 256); sw(Wh2, wh2s, 128, 128);
  sw(Wv1, wv1s, 384, 384); sw(Wv2, wv2s, 128, 128);
  sw(Wt1, wt1s, 128, 128); sw(Wt2, wt2s, 128, 128);

  zero_kernel<<<2048, 256, 0, stream>>>(agg, v0, (long long)NPAD*128, (long long)NN*384);

  node_f_kernel<<<196, 256, 0, stream>>>(species, was, ba, f32f, f16f);

  edge_kernel<<<1024, 256, 0, stream>>>(eidx32, edge_attr, edge_vec, f32f, f16f,
      wb1s, bb1, wb2s, bb2, ws1s, bs1, ws2s, bs2, wv1s, bv1, wv2s, bv2,
      ea, agg, v0);

  node_h_kernel<<<196, 256, 0, stream>>>(f16f, agg, tin, Brff,
      wh1s, bh1, wh2s, bh2, wt1s, bt1, wt2s, bt2, h0);
}
